// DWSABlock_20212116095365
// MI455X (gfx1250) — compile-verified, matrix operand order checked by reading, not run on hardware
//
#include <hip/hip_runtime.h>

// ---------------------------------------------------------------------------
// DWSA block for MI455X (gfx1250): fused windowed attention, bf16 WMMA path.
// Grid: 4096 blocks (one 8x8 window), 256 threads = 8 wave32s.
// ---------------------------------------------------------------------------

typedef __attribute__((ext_vector_type(16))) __bf16 v16bf;
typedef __attribute__((ext_vector_type(8)))  float  v8f;

union FragB { unsigned int u[8]; v16bf v; };

#define WMMA_BF16(A, B, C) \
  __builtin_amdgcn_wmma_f32_16x16x32_bf16(false, (A), false, (B), (short)0, (C), false, false)

// K offset for VGPR j within a 16-bit A/B operand (CDNA5 ISA 7.12.2):
// VGPR 0-3: lanes0-15 K=2j,2j+1 ; lanes16-31 K=8+2j,8+2j+1 ; VGPR 4-7: +16.
__device__ __forceinline__ int kOff(int j, int lane) {
  return ((j & 4) << 2) | ((lane & 16) >> 1) | ((j & 3) << 1);
}

// Row-major MxK bf16 -> A fragment (also B fragment from column-major [n][k]).
__device__ __forceinline__ v16bf ldFrag(const __bf16* base, int ldk, int m0, int k0, int lane) {
  FragB f;
  const __bf16* row = base + (m0 + (lane & 15)) * ldk + k0;
#pragma unroll
  for (int j = 0; j < 8; ++j)
    f.u[j] = *(const unsigned int*)(row + kOff(j, lane));
  return f.v;
}

// Same, but K >= kmax reads as zero (K-dimension zero padding).
__device__ __forceinline__ v16bf ldFragKPad(const __bf16* base, int ldk, int m0, int kmax, int lane) {
  FragB f;
  const __bf16* row = base + (m0 + (lane & 15)) * ldk;
#pragma unroll
  for (int j = 0; j < 8; ++j) {
    int k = kOff(j, lane);
    f.u[j] = (k < kmax) ? *(const unsigned int*)(row + k) : 0u;
  }
  return f.v;
}

// Same, but rows m >= mmax read as zero (M-dimension zero padding).
__device__ __forceinline__ v16bf ldFragMPad(const __bf16* base, int ldk, int mmax, int k0, int lane) {
  FragB f;
  int m = lane & 15;
  const __bf16* row = base + m * ldk + k0;
#pragma unroll
  for (int j = 0; j < 8; ++j)
    f.u[j] = (m < mmax) ? *(const unsigned int*)(row + kOff(j, lane)) : 0u;
  return f.v;
}

// ----------------------- workspace layout (bytes) ---------------------------
#define WS_WQK   0      // bf16 16x32 dense(block-diag) w_qk
#define WS_WQ    1024   // bf16 16x32 (cols 16..31 zero) w_q
#define WS_WK    2048   // bf16 16x32 (cols 16..31 zero) w_k
#define WS_WV    3072   // bf16 16x32 dense(block-diag) w_v
#define WS_WO    4096   // bf16 32x32 dense(block-diag, cols 16..31 zero) w_o
#define WS_WIN   6144   // bf16 32x128 w_in
#define WS_WOUT  14336  // bf16 128x32 w_out
#define WS_INVIN 22528  // f32[128] bn_in scale
#define WS_BETIN 23040  // f32[128] bn_in shift
#define WS_SC    23552  // f32[128] clip(alpha)*bn_out scale
#define WS_SB    24064  // f32[128] clip(alpha)*bn_out shift
#define WS_BQ    24576  // f32[16]
#define WS_BK    24640  // f32[16]
#define WS_BV    24704  // f32[16]
#define WS_BO    24768  // f32[32]

__global__ __launch_bounds__(256) void dwsa_prep(
    const float* __restrict__ bn_in_g, const float* __restrict__ bn_in_b,
    const float* __restrict__ bn_in_m, const float* __restrict__ bn_in_v,
    const float* __restrict__ w_in,  const float* __restrict__ w_qk,
    const float* __restrict__ w_q,   const float* __restrict__ b_q,
    const float* __restrict__ w_k,   const float* __restrict__ b_k,
    const float* __restrict__ w_v,   const float* __restrict__ b_v,
    const float* __restrict__ w_o,   const float* __restrict__ b_o,
    const float* __restrict__ w_out,
    const float* __restrict__ bn_out_g, const float* __restrict__ bn_out_b,
    const float* __restrict__ bn_out_m, const float* __restrict__ bn_out_v,
    const float* __restrict__ alpha, char* __restrict__ ws)
{
  __bf16* WQK  = (__bf16*)(ws + WS_WQK);
  __bf16* WQ   = (__bf16*)(ws + WS_WQ);
  __bf16* WK   = (__bf16*)(ws + WS_WK);
  __bf16* WV   = (__bf16*)(ws + WS_WV);
  __bf16* WO   = (__bf16*)(ws + WS_WO);
  __bf16* WIN  = (__bf16*)(ws + WS_WIN);
  __bf16* WOUT = (__bf16*)(ws + WS_WOUT);
  float* INVIN = (float*)(ws + WS_INVIN);
  float* BETIN = (float*)(ws + WS_BETIN);
  float* SC    = (float*)(ws + WS_SC);
  float* SB    = (float*)(ws + WS_SB);
  float* BQ    = (float*)(ws + WS_BQ);
  float* BK    = (float*)(ws + WS_BK);
  float* BV    = (float*)(ws + WS_BV);
  float* BO    = (float*)(ws + WS_BO);

  int tid = threadIdx.x;
  if (tid < 128) {
    float inv = bn_in_g[tid] * rsqrtf(bn_in_v[tid] + 1e-5f);
    INVIN[tid] = inv;
    BETIN[tid] = bn_in_b[tid] - bn_in_m[tid] * inv;
    float a    = fminf(fmaxf(alpha[tid], 0.f), 1.f);
    float invo = bn_out_g[tid] * rsqrtf(bn_out_v[tid] + 1e-5f);
    SC[tid] = a * invo;
    SB[tid] = a * (bn_out_b[tid] - bn_out_m[tid] * invo);
  }
  if (tid < 16) { BQ[tid] = b_q[tid]; BK[tid] = b_k[tid]; BV[tid] = b_v[tid]; }
  if (tid < 32) { BO[tid] = b_o[tid]; }

  // Grouped 16x8 (4 groups) -> dense block-diagonal 16x32; w_q/w_k padded to K=32.
  for (int i = tid; i < 512; i += 256) {
    int o = i >> 5, k = i & 31;
    WQK[i] = (__bf16)(((k >> 3) == (o >> 2)) ? w_qk[o * 8 + (k & 7)] : 0.f);
    WV [i] = (__bf16)(((k >> 3) == (o >> 2)) ? w_v [o * 8 + (k & 7)] : 0.f);
    WQ [i] = (__bf16)((k < 16) ? w_q[o * 16 + k] : 0.f);
    WK [i] = (__bf16)((k < 16) ? w_k[o * 16 + k] : 0.f);
  }
  // Grouped 32x4 (4 groups) -> dense block-diagonal 32x16, padded to K=32.
  for (int i = tid; i < 1024; i += 256) {
    int o = i >> 5, k = i & 31;
    WO[i] = (__bf16)((k < 16 && (k >> 2) == (o >> 3)) ? w_o[o * 4 + (k & 3)] : 0.f);
  }
  for (int i = tid; i < 4096; i += 256) {
    WIN[i]  = (__bf16)w_in[i];    // 32x128 row-major
    WOUT[i] = (__bf16)w_out[i];   // 128x32 row-major
  }
}

// -------------------------------- main --------------------------------------
__global__ __launch_bounds__(256) void dwsa_main(const float* __restrict__ x,
                                                 float* __restrict__ out,
                                                 const char* __restrict__ ws)
{
  // LDS (43 KB), lifetimes overlapped:
  //   region0 [0,32768):    XN bf16[64][128] (stage1..G1)  ->  S f32[2][64][64]
  //                         (G4..softmax) -> P bf16[2][64][64] (in-place, reg-buffered)
  //   region1 [32768,43008): XF bf16[64][32] (G1..G2) -> QCM/KCM bf16[64][16]x2
  //                          (G3..G4) -> OCM bf16[64][32] (G6..G7)
  //                          BASE bf16[64][32] (G2..G3) -> OUTM bf16[64][16] (G5..G6)
  //                          VRM bf16[16][64] (G2..G5)
  __shared__ char smem[43008];
  float*  S    = (float*)(smem);
  __bf16* XN   = (__bf16*)(smem);
  __bf16* P    = (__bf16*)(smem);
  char* r1 = smem + 32768;
  __bf16* XF   = (__bf16*)(r1);
  __bf16* QCM  = (__bf16*)(r1);
  __bf16* KCM  = (__bf16*)(r1 + 2048);
  __bf16* OCM  = (__bf16*)(r1);
  __bf16* BASE = (__bf16*)(r1 + 4096);
  __bf16* OUTM = (__bf16*)(r1 + 4096);
  __bf16* VRM  = (__bf16*)(r1 + 8192);

  const __bf16* WQK  = (const __bf16*)(ws + WS_WQK);
  const __bf16* WQp  = (const __bf16*)(ws + WS_WQ);
  const __bf16* WKp  = (const __bf16*)(ws + WS_WK);
  const __bf16* WV   = (const __bf16*)(ws + WS_WV);
  const __bf16* WOp  = (const __bf16*)(ws + WS_WO);
  const __bf16* WIN  = (const __bf16*)(ws + WS_WIN);
  const __bf16* WOUT = (const __bf16*)(ws + WS_WOUT);
  const float* INVIN = (const float*)(ws + WS_INVIN);
  const float* BETIN = (const float*)(ws + WS_BETIN);
  const float* SC    = (const float*)(ws + WS_SC);
  const float* SB    = (const float*)(ws + WS_SB);
  const float* BQ    = (const float*)(ws + WS_BQ);
  const float* BK    = (const float*)(ws + WS_BK);
  const float* BV    = (const float*)(ws + WS_BV);
  const float* BO    = (const float*)(ws + WS_BO);

  const int tid = threadIdx.x, lane = tid & 31, wave = tid >> 5;
  const int blk = blockIdx.x;
  const int b  = blk >> 8;
  const int nh = (blk >> 4) & 15;
  const int nw = blk & 15;
  // element offset of window origin within x: b*128*16384 + h0*128 + w0
  const long base0 = ((long)b << 21) + ((long)nh << 10) + ((long)nw << 3);
  const int mh = (lane & 16) >> 1;   // +8 row offset for upper lane half in C/D

  // ---- Stage 1: load x window, apply BN_in, store bf16 token-major [t][c] ----
#pragma unroll
  for (int i = 0; i < 8; ++i) {
    int lin = tid + (i << 8);            // 0..2047 : (c, r, half)
    int c = lin >> 4, sub = lin & 15;
    int r = sub >> 1, half = (sub & 1) << 2;
    const float4 val = *(const float4*)(x + base0 + ((long)c << 14) + (r << 7) + half);
    float inv = INVIN[c], bet = BETIN[c];
    int t = (r << 3) + half;
    XN[(t + 0) * 128 + c] = (__bf16)(val.x * inv + bet);
    XN[(t + 1) * 128 + c] = (__bf16)(val.y * inv + bet);
    XN[(t + 2) * 128 + c] = (__bf16)(val.z * inv + bet);
    XN[(t + 3) * 128 + c] = (__bf16)(val.w * inv + bet);
  }
  __syncthreads();

  // ---- G1: x_red(32x64) = w_in(32x128) @ xn(128x64), 1 tile per wave ----
  {
    int mi = wave & 1, ni = wave >> 1;
    v8f acc = {};
#pragma unroll
    for (int kk = 0; kk < 4; ++kk) {
      v16bf a  = ldFrag(WIN, 128, mi * 16, kk * 32, lane);
      v16bf bb = ldFrag(XN,  128, ni * 16, kk * 32, lane);
      acc = WMMA_BF16(a, bb, acc);
    }
    int t = ni * 16 + (lane & 15);
#pragma unroll
    for (int v2 = 0; v2 < 8; ++v2)
      XF[t * 32 + mi * 16 + v2 + mh] = (__bf16)acc[v2];
  }
  __syncthreads();

  // ---- G2: base = Wqk_d @ xf (waves 0-3) ; v = Wv_d @ xf + b_v (waves 4-7) ----
  {
    int ni = wave & 3;
    v16bf a  = ldFrag((wave < 4) ? WQK : WV, 32, 0, 0, lane);
    v16bf bb = ldFrag(XF, 32, ni * 16, 0, lane);
    v8f acc = {};
    acc = WMMA_BF16(a, bb, acc);
    int t = ni * 16 + (lane & 15);
    if (wave < 4) {
#pragma unroll
      for (int v2 = 0; v2 < 8; ++v2) BASE[t * 32 + v2 + mh] = (__bf16)acc[v2];
      if (lane < 16) {
#pragma unroll
        for (int z = 16; z < 32; ++z) BASE[t * 32 + z] = (__bf16)0.f;  // K-pad
      }
    } else {
#pragma unroll
      for (int v2 = 0; v2 < 8; ++v2) {
        int m = v2 + mh;
        VRM[m * 64 + t] = (__bf16)(acc[v2] + BV[m]);
      }
    }
  }
  __syncthreads();

  // ---- G3: q = Wq @ base + b_q (waves 0-3, scale folded) ; k likewise (4-7) ----
  {
    int ni = wave & 3;
    v16bf a  = ldFrag((wave < 4) ? WQp : WKp, 32, 0, 0, lane);
    v16bf bb = ldFrag(BASE, 32, ni * 16, 0, lane);
    v8f acc = {};
    acc = WMMA_BF16(a, bb, acc);
    int t = ni * 16 + (lane & 15);
    if (wave < 4) {
#pragma unroll
      for (int v2 = 0; v2 < 8; ++v2) {
        int m = v2 + mh;
        QCM[t * 16 + m] = (__bf16)((acc[v2] + BQ[m]) * 0.35355339059f);  // hd^-0.5
      }
    } else {
#pragma unroll
      for (int v2 = 0; v2 < 8; ++v2) {
        int m = v2 + mh;
        KCM[t * 16 + m] = (__bf16)(acc[v2] + BK[m]);
      }
    }
  }
  __syncthreads();

  // ---- G4: S_h(64x64) = q_h^T @ k_h, K=8 zero-padded to 32. 4 tiles per wave ----
  {
    int h = wave >> 2, mi = wave & 3;
    v16bf a = ldFragKPad(QCM + h * 8, 16, mi * 16, 8, lane);
#pragma unroll
    for (int ni = 0; ni < 4; ++ni) {
      v16bf bb = ldFragKPad(KCM + h * 8, 16, ni * 16, 8, lane);
      v8f acc = {};
      acc = WMMA_BF16(a, bb, acc);
      int nk = ni * 16 + (lane & 15);
#pragma unroll
      for (int v2 = 0; v2 < 8; ++v2)
        S[h * 4096 + (mi * 16 + v2 + mh) * 64 + nk] = acc[v2];
    }
  }
  __syncthreads();

  // ---- Softmax over keys: 128 rows (h, n). Register-buffered, P overwrites S ----
  {
    float vbuf[64];
    float rinv = 0.f;
    if (tid < 128) {
      const float* rowp = S + tid * 64;
      float mx = -3.4e38f;
#pragma unroll
      for (int m = 0; m < 64; ++m) { vbuf[m] = rowp[m]; mx = fmaxf(mx, vbuf[m]); }
      float sum = 0.f;
#pragma unroll
      for (int m = 0; m < 64; ++m) { vbuf[m] = __expf(vbuf[m] - mx); sum += vbuf[m]; }
      rinv = 1.f / sum;
    }
    __syncthreads();   // everyone: reads done before bf16 rows alias float rows
    if (tid < 128) {
      __bf16* prow = P + tid * 64;
#pragma unroll
      for (int m = 0; m < 64; ++m) prow[m] = (__bf16)(vbuf[m] * rinv);
    }
  }
  __syncthreads();

  // ---- G5: out_h(8x64) = v_h(8x64) @ P_h^T(64x64), M-pad to 16, K=64 (2 steps) ----
  {
    int h = wave >> 2, ni = wave & 3;
    v8f acc = {};
#pragma unroll
    for (int ks = 0; ks < 2; ++ks) {
      v16bf a  = ldFragMPad(VRM + h * 8 * 64, 64, 8, ks * 32, lane);
      v16bf bb = ldFrag(P + h * 4096, 64, ni * 16, ks * 32, lane);
      acc = WMMA_BF16(a, bb, acc);
    }
    if (lane < 16) {
      int t = ni * 16 + lane;
#pragma unroll
      for (int v2 = 0; v2 < 8; ++v2)
        OUTM[t * 16 + h * 8 + v2] = (__bf16)acc[v2];
    }
  }
  __syncthreads();

  // ---- G6: o(32x64) = Wo_d(32x16 pad32) @ outm(16x64) + b_o ----
  {
    int mi = wave & 1, ni = wave >> 1;
    v16bf a  = ldFrag(WOp, 32, mi * 16, 0, lane);
    v16bf bb = ldFragKPad(OUTM, 16, ni * 16, 16, lane);
    v8f acc = {};
    acc = WMMA_BF16(a, bb, acc);
    int t = ni * 16 + (lane & 15);
#pragma unroll
    for (int v2 = 0; v2 < 8; ++v2) {
      int m = mi * 16 + v2 + mh;
      OCM[t * 32 + m] = (__bf16)(acc[v2] + BO[m]);
    }
  }
  __syncthreads();

  // ---- G7: y = x + sc[c]*(w_out @ o) + sb[c], fused BN_out + residual + store ----
  {
    int mi = wave;                               // channel tile 16*wave
    v16bf a = ldFrag(WOUT, 32, mi * 16, 0, lane);
#pragma unroll
    for (int ni = 0; ni < 4; ++ni) {
      v16bf bb = ldFrag(OCM, 32, ni * 16, 0, lane);
      v8f acc = {};
      acc = WMMA_BF16(a, bb, acc);
      int t = ni * 16 + (lane & 15);
      int r = t >> 3, cw = t & 7;
#pragma unroll
      for (int v2 = 0; v2 < 8; ++v2) {
        int c = mi * 16 + v2 + mh;
        long gi = base0 + ((long)c << 14) + (r << 7) + cw;
        out[gi] = x[gi] + acc[v2] * SC[c] + SB[c];
      }
    }
  }
}

// ---------------------------------------------------------------------------
extern "C" void kernel_launch(void* const* d_in, const int* in_sizes, int n_in,
                              void* d_out, int out_size, void* d_ws, size_t ws_size,
                              hipStream_t stream) {
  (void)in_sizes; (void)n_in; (void)out_size; (void)ws_size;
  dwsa_prep<<<1, 256, 0, stream>>>(
      (const float*)d_in[1],  (const float*)d_in[2],  (const float*)d_in[3],
      (const float*)d_in[4],  (const float*)d_in[5],  (const float*)d_in[6],
      (const float*)d_in[7],  (const float*)d_in[8],  (const float*)d_in[9],
      (const float*)d_in[10], (const float*)d_in[11], (const float*)d_in[12],
      (const float*)d_in[13], (const float*)d_in[14], (const float*)d_in[15],
      (const float*)d_in[16], (const float*)d_in[17], (const float*)d_in[18],
      (const float*)d_in[19], (const float*)d_in[20], (char*)d_ws);
  dwsa_main<<<4096, 256, 0, stream>>>((const float*)d_in[0], (float*)d_out,
                                      (const char*)d_ws);
}
